// AFP_4234837754146
// MI455X (gfx1250) — compile-verified, matrix operand order checked by reading, not run on hardware
//
#include <hip/hip_runtime.h>
#include <math.h>

// ---------------- problem constants (match reference) ----------------
constexpr int NN  = 100000;   // nodes
constexpr int NE  = 400000;   // edges
constexpr int NG  = 4096;     // graphs
constexpr int HH  = 256;      // hidden
constexpr int H3  = 768;      // 3*H
constexpr int INC = 39;       // input channels
constexpr int ED  = 10;       // edge feature dim
constexpr int NT  = 2;        // pooling timesteps
#define LEAKY 0.01f

// ---------------- types for WMMA ----------------
typedef __bf16 bf16;
typedef __attribute__((ext_vector_type(16))) __bf16 v16bf;
typedef __attribute__((ext_vector_type(8)))  __bf16 v8bf;
typedef __attribute__((ext_vector_type(8)))  float  v8f;

__device__ __forceinline__ v8f wmma_bf16(v16bf a, v16bf b, v8f c) {
  return __builtin_amdgcn_wmma_f32_16x16x32_bf16(false, a, false, b, (short)0, c,
                                                 false, false);
}

// per-lane 16-bit fragment: K-groups {half*8..half*8+7} and {16+half*8..}
__device__ __forceinline__ v16bf load_frag(const bf16* row_k, int half) {
  v8bf lo = *(const v8bf*)(row_k + half * 8);
  v8bf hi = *(const v8bf*)(row_k + 16 + half * 8);
  v16bf r;
#pragma unroll
  for (int i = 0; i < 8; ++i) { r[i] = lo[i]; r[8 + i] = hi[i]; }
  return r;
}

__device__ __forceinline__ void atomAddF(float* p, float v) { unsafeAtomicAdd(p, v); }

// monotone float<->uint mapping for atomicMax over signed floats
__device__ __forceinline__ unsigned fmap(float f) {
  unsigned u = __float_as_uint(f);
  return (u & 0x80000000u) ? ~u : (u | 0x80000000u);
}
__device__ __forceinline__ float funmap(unsigned u) {
  return __uint_as_float((u & 0x80000000u) ? (u & 0x7fffffffu) : ~u);
}

// ============================================================================
// WMMA GEMM:  C[M,Nn] = A[M,K](bf16) * B[Nn,K](bf16)^T  (+bias)
// 256 threads = 8 waves, each wave a 32x32 tile; block macro-tile 128x64.
// SCATTER epilogue: atomicAdd C[rdst[row]*Nn + col] += acc*rscale[row]
// ============================================================================
template <bool SCATTER>
__global__ __launch_bounds__(256) void k_gemm_bf16(
    const bf16* __restrict__ A, const bf16* __restrict__ B,
    const float* __restrict__ bias, float* __restrict__ C, int M, int Nn, int K,
    const int* __restrict__ rdst, const float* __restrict__ rscale) {
  const int lane = threadIdx.x & 31;
  const int wave = threadIdx.x >> 5;
  const int half = lane >> 4;
  const int l16  = lane & 15;
  const int wm = wave & 3, wn = wave >> 2;  // 4 (M) x 2 (N) waves
  const int m0 = blockIdx.y * 128 + wm * 32;
  const int n0 = blockIdx.x * 64 + wn * 32;
  if (m0 >= M || n0 >= Nn) return;  // wave-uniform; EXEC stays all-ones

  v8f a00 = {}, a01 = {}, a10 = {}, a11 = {};
  const bf16* Ar0 = A + (size_t)(m0 + l16) * K;
  const bf16* Ar1 = A + (size_t)(m0 + 16 + l16) * K;
  const bf16* Br0 = B + (size_t)(n0 + l16) * K;
  const bf16* Br1 = B + (size_t)(n0 + 16 + l16) * K;
  for (int k = 0; k < K; k += 32) {
    v16bf fa0 = load_frag(Ar0 + k, half);
    v16bf fa1 = load_frag(Ar1 + k, half);
    v16bf fb0 = load_frag(Br0 + k, half);
    v16bf fb1 = load_frag(Br1 + k, half);
    a00 = wmma_bf16(fa0, fb0, a00);
    a01 = wmma_bf16(fa0, fb1, a01);
    a10 = wmma_bf16(fa1, fb0, a10);
    a11 = wmma_bf16(fa1, fb1, a11);
  }
#pragma unroll
  for (int mi = 0; mi < 2; ++mi) {
#pragma unroll
    for (int ni = 0; ni < 2; ++ni) {
      v8f acc = (mi == 0) ? (ni == 0 ? a00 : a01) : (ni == 0 ? a10 : a11);
      const int col = n0 + ni * 16 + l16;
#pragma unroll
      for (int v = 0; v < 8; ++v) {
        const int row = m0 + mi * 16 + half * 8 + v;
        float val = acc[v];
        if constexpr (SCATTER) {
          atomAddF(&C[(size_t)rdst[row] * Nn + col], val * rscale[row]);
        } else {
          if (bias) val += bias[col];
          C[(size_t)row * Nn + col] = val;
        }
      }
    }
  }
}

// ============================================================================
// small kernels
// ============================================================================
__global__ void k_cvt_bf16(const float* __restrict__ in, bf16* __restrict__ out,
                           long long n) {
  long long i = (long long)blockIdx.x * 256 + threadIdx.x;
  if (i < n) out[i] = (bf16)in[i];
}

__global__ void k_fill_u32(unsigned* __restrict__ p, unsigned v, long long n) {
  long long i = (long long)blockIdx.x * 256 + threadIdx.x;
  if (i < n) p[i] = v;
}

__global__ void k_copy_f32(const float* __restrict__ in, float* __restrict__ out,
                           long long n) {
  long long i = (long long)blockIdx.x * 256 + threadIdx.x;
  if (i < n) out[i] = in[i];
}

// pack first HH columns of ge_lin1_w [HH, HH+ED] into bf16 [HH, HH]
__global__ void k_pack_w1x(const float* __restrict__ w, bf16* __restrict__ out) {
  int i = blockIdx.x * 256 + threadIdx.x;  // HH*HH total
  int r = i >> 8, c = i & 255;
  out[i] = (bf16)w[r * (HH + ED) + c];
}

// x1 = leaky(x @ lin1_w^T + b), K=39 scalar; writes f32 + bf16
__global__ void k_lin1(const float* __restrict__ x, const float* __restrict__ w,
                       const float* __restrict__ b, float* __restrict__ xo,
                       bf16* __restrict__ xobf) {
  __shared__ float xs[INC];
  const int n = blockIdx.x, c = threadIdx.x;
  if (c < INC) xs[c] = x[(size_t)n * INC + c];
  __syncthreads();
  float acc = b[c];
#pragma unroll
  for (int j = 0; j < INC; ++j) acc += xs[j] * w[c * INC + j];
  acc = acc > 0.f ? acc : LEAKY * acc;
  xo[(size_t)n * HH + c] = acc;
  xobf[(size_t)n * HH + c] = (bf16)acc;
}

// out[r] = dot(X[r,:HH], v) + (bias?bias[0]:0); one wave per row
__global__ void k_rowdot(const float* __restrict__ X, const float* __restrict__ v,
                         float* __restrict__ out, const float* __restrict__ bias) {
  const int r = blockIdx.x, lane = threadIdx.x;
  float s = 0.f;
#pragma unroll
  for (int c = lane; c < HH; c += 32) s += X[(size_t)r * HH + c] * v[c];
#pragma unroll
  for (int off = 16; off; off >>= 1) s += __shfl_xor(s, off, 32);
  if (lane == 0) out[r] = s + (bias ? bias[0] : 0.f);
}

// GATEConv message: m = leaky(t[src] + eattr @ W1e^T); writes bf16 m and
// ml[e] = dot(m, ge_att_l)
__global__ void k_build_m(const float* __restrict__ t, const float* __restrict__ eattr,
                          const float* __restrict__ w,  // ge_lin1_w [HH, HH+ED]
                          const float* __restrict__ attl, const int* __restrict__ src,
                          bf16* __restrict__ mbf, float* __restrict__ ml) {
  __shared__ float es[ED];
  __shared__ float red[256];
  const int e = blockIdx.x, c = threadIdx.x;
  if (c < ED) es[c] = eattr[(size_t)e * ED + c];
  __syncthreads();
  const int s = src[e];
  float acc = t[(size_t)s * HH + c];
#pragma unroll
  for (int j = 0; j < ED; ++j) acc += es[j] * w[c * (HH + ED) + HH + j];
  acc = acc > 0.f ? acc : LEAKY * acc;
  mbf[(size_t)e * HH + c] = (bf16)acc;
  red[c] = acc * attl[c];
  __syncthreads();
  for (int st = 128; st; st >>= 1) {
    if (c < st) red[c] += red[c + st];
    __syncthreads();
  }
  if (c == 0) ml[e] = red[0];
}

// a[e] = leaky((src? s[src[e]] : s[e]) + sd[dst[e]]); track per-group max
__global__ void k_alpha_max(const float* __restrict__ ssrc, const int* __restrict__ src,
                            const float* __restrict__ sdst, const int* __restrict__ dst,
                            int E, float* __restrict__ aval, unsigned* __restrict__ amax) {
  int e = blockIdx.x * 256 + threadIdx.x;
  if (e >= E) return;
  float a = (src ? ssrc[src[e]] : ssrc[e]) + sdst[dst[e]];
  a = a > 0.f ? a : LEAKY * a;
  aval[e] = a;
  atomicMax(&amax[dst[e]], fmap(a));
}

__global__ void k_alpha_sum(const float* __restrict__ aval, const int* __restrict__ dst,
                            const unsigned* __restrict__ amax, float* __restrict__ asum,
                            float* __restrict__ aexp, int E) {
  int e = blockIdx.x * 256 + threadIdx.x;
  if (e >= E) return;
  int d = dst[e];
  float ex = expf(aval[e] - funmap(amax[d]));
  aexp[e] = ex;
  atomAddF(&asum[d], ex);
}

__global__ void k_alpha_wgt(const float* __restrict__ aexp, const float* __restrict__ asum,
                            const int* __restrict__ dst, float* __restrict__ wgt, int E) {
  int e = blockIdx.x * 256 + threadIdx.x;
  if (e >= E) return;
  wgt[e] = aexp[e] / (asum[dst[e]] + 1e-16f);
}

// h[dst[e]] += hs[src?src[e]:e] * wgt[e]
__global__ void k_gat_scatter(const float* __restrict__ hs, const int* __restrict__ src,
                              const int* __restrict__ dst, const float* __restrict__ wgt,
                              float* __restrict__ hout) {
  const int e = blockIdx.x, c = threadIdx.x;
  const int s = src ? src[e] : e;
  atomAddF(&hout[(size_t)dst[e] * HH + c], hs[(size_t)s * HH + c] * wgt[e]);
}

// h = elu(h + bias); also write bf16 copy
__global__ void k_bias_elu_bf(float* __restrict__ h, const float* __restrict__ bias,
                              bf16* __restrict__ hbf, long long n) {
  long long i = (long long)blockIdx.x * 256 + threadIdx.x;
  if (i >= n) return;
  float v = h[i] + bias[(int)(i & (HH - 1))];
  v = v > 0.f ? v : (expf(v) - 1.f);
  h[i] = v;
  hbf[i] = (bf16)v;
}

__global__ void k_relu_bf(float* __restrict__ p, bf16* __restrict__ pbf, long long n) {
  long long i = (long long)blockIdx.x * 256 + threadIdx.x;
  if (i >= n) return;
  float v = fmaxf(p[i], 0.f);
  p[i] = v;
  pbf[i] = (bf16)v;
}

// pooled[batch[n]] += x[n]
__global__ void k_pool(const float* __restrict__ x, const int* __restrict__ batch,
                       float* __restrict__ out) {
  const int n = blockIdx.x, c = threadIdx.x;
  atomAddF(&out[(size_t)batch[n] * HH + c], x[(size_t)n * HH + c]);
}

// GRU gates (biases already in gi/gh); out = relu(gru); safe in-place vs hp
__global__ void k_gru(const float* __restrict__ gi, const float* __restrict__ gh,
                      const float* __restrict__ hp, float* __restrict__ xo,
                      bf16* __restrict__ xobf, int M) {
  long long i = (long long)blockIdx.x * 256 + threadIdx.x;
  if (i >= (long long)M * HH) return;
  const int r = (int)(i >> 8), c = (int)(i & 255);
  const size_t b = (size_t)r * H3;
  float ir = gi[b + c], iz = gi[b + HH + c], in = gi[b + 2 * HH + c];
  float hr = gh[b + c], hz = gh[b + HH + c], hn = gh[b + 2 * HH + c];
  float rr = 1.f / (1.f + expf(-(ir + hr)));
  float zz = 1.f / (1.f + expf(-(iz + hz)));
  float nn = tanhf(in + rr * hn);
  float h0 = hp[(size_t)r * HH + c];
  float o = fmaxf((1.f - zz) * nn + zz * h0, 0.f);
  xo[(size_t)r * HH + c] = o;
  if (xobf) xobf[(size_t)r * HH + c] = (bf16)o;
}

// ============================================================================
// host
// ============================================================================
static void gemm(hipStream_t s, const bf16* A, const bf16* B, const float* bias,
                 float* C, int M, int Nn, int K) {
  dim3 g((Nn + 63) / 64, (M + 127) / 128);
  k_gemm_bf16<false><<<g, 256, 0, s>>>(A, B, bias, C, M, Nn, K, nullptr, nullptr);
}
static void gemm_scatter(hipStream_t s, const bf16* A, const bf16* B, float* C, int M,
                         int Nn, int K, const int* rdst, const float* rscale) {
  dim3 g((Nn + 63) / 64, (M + 127) / 128);
  k_gemm_bf16<true><<<g, 256, 0, s>>>(A, B, nullptr, C, M, Nn, K, rdst, rscale);
}
static inline long long gb(long long n) { return (n + 255) / 256; }

extern "C" void kernel_launch(void* const* d_in, const int* in_sizes, int n_in,
                              void* d_out, int out_size, void* d_ws, size_t ws_size,
                              hipStream_t stream) {
  (void)in_sizes; (void)n_in; (void)out_size; (void)ws_size;
  const float* x          = (const float*)d_in[0];
  const int*   eidx       = (const int*)d_in[1];
  const float* eattr      = (const float*)d_in[2];
  const int*   batch      = (const int*)d_in[3];
  const float* lin1_w     = (const float*)d_in[4];
  const float* lin1_b     = (const float*)d_in[5];
  const float* ge_att_l   = (const float*)d_in[6];
  const float* ge_att_r   = (const float*)d_in[7];
  const float* ge_lin1_w  = (const float*)d_in[8];
  const float* ge_lin2_w  = (const float*)d_in[9];
  const float* ge_bias    = (const float*)d_in[10];
  const float* gat_lin_w  = (const float*)d_in[11];
  const float* gat_att_src= (const float*)d_in[12];
  const float* gat_att_dst= (const float*)d_in[13];
  const float* gat_bias   = (const float*)d_in[14];
  const float* gru_wih    = (const float*)d_in[15];
  const float* gru_whh    = (const float*)d_in[16];
  const float* gru_bih    = (const float*)d_in[17];
  const float* gru_bhh    = (const float*)d_in[18];
  const float* mol_lin_w  = (const float*)d_in[19];
  const float* mol_att_src= (const float*)d_in[20];
  const float* mol_att_dst= (const float*)d_in[21];
  const float* mol_bias   = (const float*)d_in[22];
  const float* mol_wih    = (const float*)d_in[23];
  const float* mol_whh    = (const float*)d_in[24];
  const float* mol_bih    = (const float*)d_in[25];
  const float* mol_bhh    = (const float*)d_in[26];
  const float* lin2_w     = (const float*)d_in[27];
  const float* lin2_b     = (const float*)d_in[28];
  const int* src = eidx;
  const int* dst = eidx + NE;

  // ---- bump allocator over d_ws ----
  char* p = (char*)d_ws;
  auto alloc = [&](size_t bytes) -> void* {
    void* r = (void*)p;
    p += (bytes + 255) & ~(size_t)255;
    return r;
  };
  float* xf    = (float*)alloc((size_t)NN * HH * 4);
  bf16*  xbf   = (bf16*)alloc((size_t)NN * HH * 2);
  float* hf    = (float*)alloc((size_t)NN * HH * 4);
  bf16*  hbf   = (bf16*)alloc((size_t)NN * HH * 2);
  float* hs    = (float*)alloc((size_t)NN * HH * 4);   // t / hs / hs_mol
  bf16*  mbf   = (bf16*)alloc((size_t)NE * HH * 2);
  float* gi    = (float*)alloc((size_t)NN * H3 * 4);
  float* gh    = (float*)alloc((size_t)NN * H3 * 4);
  float* ml    = (float*)alloc((size_t)NE * 4);
  float* aval  = (float*)alloc((size_t)NE * 4);
  float* aexp  = (float*)alloc((size_t)NE * 4);
  float* wgt   = (float*)alloc((size_t)NE * 4);
  float* nas   = (float*)alloc((size_t)NN * 4);
  float* nad   = (float*)alloc((size_t)NN * 4);
  unsigned* amax = (unsigned*)alloc((size_t)NN * 4);
  float* asum  = (float*)alloc((size_t)NN * 4);
  float* outg  = (float*)alloc((size_t)NG * HH * 4);
  bf16*  outgbf= (bf16*)alloc((size_t)NG * HH * 2);
  float* hg    = (float*)alloc((size_t)NG * HH * 4);
  bf16*  hgbf  = (bf16*)alloc((size_t)NG * HH * 2);
  float* hd    = (float*)alloc((size_t)NG * HH * 4);
  float* gig   = (float*)alloc((size_t)NG * H3 * 4);
  float* ghg   = (float*)alloc((size_t)NG * H3 * 4);
  float* adg   = (float*)alloc((size_t)NG * 4);
  bf16* w1x_bf    = (bf16*)alloc((size_t)HH * HH * 2);
  bf16* w2_bf     = (bf16*)alloc((size_t)HH * HH * 2);
  bf16* gatw_bf   = (bf16*)alloc((size_t)2 * HH * HH * 2);
  bf16* wih_bf    = (bf16*)alloc((size_t)3 * H3 * HH * 2);
  bf16* whh_bf    = (bf16*)alloc((size_t)3 * H3 * HH * 2);
  bf16* molw_bf   = (bf16*)alloc((size_t)HH * HH * 2);
  bf16* molwih_bf = (bf16*)alloc((size_t)H3 * HH * 2);
  bf16* molwhh_bf = (bf16*)alloc((size_t)H3 * HH * 2);

  // ---- weight conversions (once per call; small) ----
  k_pack_w1x<<<HH * HH / 256, 256, 0, stream>>>(ge_lin1_w, w1x_bf);
  k_cvt_bf16<<<gb((long long)HH * HH), 256, 0, stream>>>(ge_lin2_w, w2_bf, (long long)HH * HH);
  k_cvt_bf16<<<gb(2LL * HH * HH), 256, 0, stream>>>(gat_lin_w, gatw_bf, 2LL * HH * HH);
  k_cvt_bf16<<<gb(3LL * H3 * HH), 256, 0, stream>>>(gru_wih, wih_bf, 3LL * H3 * HH);
  k_cvt_bf16<<<gb(3LL * H3 * HH), 256, 0, stream>>>(gru_whh, whh_bf, 3LL * H3 * HH);
  k_cvt_bf16<<<gb((long long)HH * HH), 256, 0, stream>>>(mol_lin_w, molw_bf, (long long)HH * HH);
  k_cvt_bf16<<<gb((long long)H3 * HH), 256, 0, stream>>>(mol_wih, molwih_bf, (long long)H3 * HH);
  k_cvt_bf16<<<gb((long long)H3 * HH), 256, 0, stream>>>(mol_whh, molwhh_bf, (long long)H3 * HH);

  // ---- input projection ----
  k_lin1<<<NN, 256, 0, stream>>>(x, lin1_w, lin1_b, xf, xbf);

  const long long NHl = (long long)NN * HH, GHl = (long long)NG * HH;

  // ---- layer 0: GATEConv ----
  gemm(stream, xbf, w1x_bf, nullptr, hs, NN, HH, HH);          // t = x @ W1x^T
  k_rowdot<<<NN, 32, 0, stream>>>(xf, ge_att_r, nad, nullptr); // (x*att_r).sum
  k_build_m<<<NE, 256, 0, stream>>>(hs, eattr, ge_lin1_w, ge_att_l, src, mbf, ml);
  k_fill_u32<<<gb(NN), 256, 0, stream>>>(amax, 0u, NN);
  k_fill_u32<<<gb(NN), 256, 0, stream>>>((unsigned*)asum, 0u, NN);
  k_alpha_max<<<gb(NE), 256, 0, stream>>>(ml, nullptr, nad, dst, NE, aval, amax);
  k_alpha_sum<<<gb(NE), 256, 0, stream>>>(aval, dst, amax, asum, aexp, NE);
  k_alpha_wgt<<<gb(NE), 256, 0, stream>>>(aexp, asum, dst, wgt, NE);
  k_fill_u32<<<gb(NHl), 256, 0, stream>>>((unsigned*)hf, 0u, NHl);
  gemm_scatter(stream, mbf, w2_bf, hf, NE, HH, HH, dst, wgt);  // h[dst]+= (m@W2)*a
  k_bias_elu_bf<<<gb(NHl), 256, 0, stream>>>(hf, ge_bias, hbf, NHl);
  gemm(stream, hbf, wih_bf, gru_bih, gi, NN, H3, HH);
  gemm(stream, xbf, whh_bf, gru_bhh, gh, NN, H3, HH);
  k_gru<<<gb(NHl), 256, 0, stream>>>(gi, gh, xf, xf, xbf, NN);

  // ---- layers 1..2: GATConv ----
  for (int l = 0; l < 2; ++l) {
    gemm(stream, xbf, gatw_bf + (size_t)l * HH * HH, nullptr, hs, NN, HH, HH);
    k_rowdot<<<NN, 32, 0, stream>>>(hs, gat_att_src + l * HH, nas, nullptr);
    k_rowdot<<<NN, 32, 0, stream>>>(hs, gat_att_dst + l * HH, nad, nullptr);
    k_fill_u32<<<gb(NN), 256, 0, stream>>>(amax, 0u, NN);
    k_fill_u32<<<gb(NN), 256, 0, stream>>>((unsigned*)asum, 0u, NN);
    k_alpha_max<<<gb(NE), 256, 0, stream>>>(nas, src, nad, dst, NE, aval, amax);
    k_alpha_sum<<<gb(NE), 256, 0, stream>>>(aval, dst, amax, asum, aexp, NE);
    k_alpha_wgt<<<gb(NE), 256, 0, stream>>>(aexp, asum, dst, wgt, NE);
    k_fill_u32<<<gb(NHl), 256, 0, stream>>>((unsigned*)hf, 0u, NHl);
    k_gat_scatter<<<NE, 256, 0, stream>>>(hs, src, dst, wgt, hf);
    k_bias_elu_bf<<<gb(NHl), 256, 0, stream>>>(hf, gat_bias + l * HH, hbf, NHl);
    gemm(stream, hbf, wih_bf + (size_t)(l + 1) * H3 * HH, gru_bih + (l + 1) * H3, gi, NN, H3, HH);
    gemm(stream, xbf, whh_bf + (size_t)(l + 1) * H3 * HH, gru_bhh + (l + 1) * H3, gh, NN, H3, HH);
    k_gru<<<gb(NHl), 256, 0, stream>>>(gi, gh, xf, xf, xbf, NN);
  }

  // ---- molecule-level attentive pooling ----
  k_fill_u32<<<gb(GHl), 256, 0, stream>>>((unsigned*)outg, 0u, GHl);
  k_pool<<<NN, 256, 0, stream>>>(xf, batch, outg);
  k_relu_bf<<<gb(GHl), 256, 0, stream>>>(outg, outgbf, GHl);
  gemm(stream, xbf, molw_bf, nullptr, hs, NN, HH, HH);         // hs_mol (loop-invariant)
  k_rowdot<<<NN, 32, 0, stream>>>(hs, mol_att_src, nas, nullptr);
  for (int t = 0; t < NT; ++t) {
    gemm(stream, outgbf, molw_bf, nullptr, hd, NG, HH, HH);
    k_rowdot<<<NG, 32, 0, stream>>>(hd, mol_att_dst, adg, nullptr);
    k_fill_u32<<<gb(NG), 256, 0, stream>>>(amax, 0u, NG);
    k_fill_u32<<<gb(NG), 256, 0, stream>>>((unsigned*)asum, 0u, NG);
    k_alpha_max<<<gb(NN), 256, 0, stream>>>(nas, nullptr, adg, batch, NN, aval, amax);
    k_alpha_sum<<<gb(NN), 256, 0, stream>>>(aval, batch, amax, asum, aexp, NN);
    k_alpha_wgt<<<gb(NN), 256, 0, stream>>>(aexp, asum, batch, wgt, NN);
    k_fill_u32<<<gb(GHl), 256, 0, stream>>>((unsigned*)hg, 0u, GHl);
    k_gat_scatter<<<NN, 256, 0, stream>>>(hs, nullptr, batch, wgt, hg);
    k_bias_elu_bf<<<gb(GHl), 256, 0, stream>>>(hg, mol_bias, hgbf, GHl);
    gemm(stream, hgbf, molwih_bf, mol_bih, gig, NG, H3, HH);
    gemm(stream, outgbf, molwhh_bf, mol_bhh, ghg, NG, H3, HH);
    k_gru<<<gb(GHl), 256, 0, stream>>>(gig, ghg, outg, outg, outgbf, NG);
  }

  // ---- head: o = fp @ lin2_w^T + b ; outputs = [o (NG), fp (NG*HH)] ----
  float* out_f = (float*)d_out;
  k_rowdot<<<NG, 32, 0, stream>>>(outg, lin2_w, out_f, lin2_b);
  k_copy_f32<<<gb(GHl), 256, 0, stream>>>(outg, out_f + NG, GHl);
}
